// RWKV_RNN_JIT_9070970929184
// MI455X (gfx1250) — matrix operand order repaired and checked by reading, not compiled, not run on hardware
//
#include <hip/hip_runtime.h>

typedef unsigned short u16;
typedef unsigned int   u32;

typedef __attribute__((ext_vector_type(16))) __bf16 bf16x16;
typedef __attribute__((ext_vector_type(8)))  float  f32x8;
typedef __attribute__((ext_vector_type(4)))  unsigned int u32x4;
typedef __attribute__((ext_vector_type(8)))  int  i32x8;
typedef __attribute__((ext_vector_type(4)))  int  i32x4;

#if defined(__has_builtin)
#if __has_builtin(__builtin_amdgcn_tensor_load_to_lds) && \
    __has_builtin(__builtin_amdgcn_s_wait_tensorcnt)
#define HAVE_TDM 1
#endif
#endif
#ifndef HAVE_TDM
#define HAVE_TDM 0
#endif

// ---------------------------------------------------------------- helpers
__device__ __forceinline__ u16 f2bf(float f) {
  union { float f; u32 u; } c; c.f = f;
  u32 r = c.u + 0x7fffu + ((c.u >> 16) & 1u);   // round-to-nearest-even
  return (u16)(r >> 16);
}
__device__ __forceinline__ float sigmoidf_(float x) {
  return 1.f / (1.f + __expf(-x));
}
// Wave-relative LDS byte offset of a __shared__ pointer: generic LDS
// addresses carry the DS offset in the low 32 bits (ISA 10.2).
__device__ __forceinline__ u32 lds_off(const void* p) {
  return (u32)(size_t)p;
}

// Block-wide layernorm over one row of D=1024 with 256 threads (4 elems/thread).
__device__ __forceinline__ void ln_row(const float* __restrict__ row,
                                       const float* __restrict__ w,
                                       const float* __restrict__ b,
                                       float* __restrict__ out,
                                       int D, float* red) {
  float vals[4];
  float s = 0.f;
  const int base = threadIdx.x;
  #pragma unroll
  for (int i = 0; i < 4; ++i) { vals[i] = row[base + i * 256]; s += vals[i]; }
  #pragma unroll
  for (int o = 16; o > 0; o >>= 1) s += __shfl_down(s, o, 32);
  const int lane = threadIdx.x & 31, wv = threadIdx.x >> 5;
  if (lane == 0) red[wv] = s;
  __syncthreads();
  if (wv == 0) {
    float v2 = (lane < 8) ? red[lane] : 0.f;
    #pragma unroll
    for (int o = 4; o > 0; o >>= 1) v2 += __shfl_down(v2, o, 32);
    if (lane == 0) red[0] = v2;
  }
  __syncthreads();
  const float mu = red[0] / (float)D;
  __syncthreads();
  float s2 = 0.f;
  #pragma unroll
  for (int i = 0; i < 4; ++i) { float d = vals[i] - mu; s2 += d * d; }
  #pragma unroll
  for (int o = 16; o > 0; o >>= 1) s2 += __shfl_down(s2, o, 32);
  if (lane == 0) red[wv] = s2;
  __syncthreads();
  if (wv == 0) {
    float v2 = (lane < 8) ? red[lane] : 0.f;
    #pragma unroll
    for (int o = 4; o > 0; o >>= 1) v2 += __shfl_down(v2, o, 32);
    if (lane == 0) red[0] = v2;
  }
  __syncthreads();
  const float rstd = rsqrtf(red[0] / (float)D + 1e-5f);
  #pragma unroll
  for (int i = 0; i < 4; ++i) {
    const int idx = base + i * 256;
    out[idx] = (vals[i] - mu) * rstd * w[idx] + b[idx];
  }
}

// ---------------------------------------------------------------- kernels
__global__ __launch_bounds__(256) void embed_ln0_kernel(
    const int* __restrict__ tokens, const float* __restrict__ emb,
    const float* __restrict__ w, const float* __restrict__ b,
    float* __restrict__ out, int D) {
  __shared__ float red[8];
  const int t = blockIdx.x;
  const float* row = emb + (size_t)tokens[t] * D;
  ln_row(row, w, b, out + (size_t)t * D, D, red);
}

__global__ __launch_bounds__(256) void ln_rows_kernel(
    const float* __restrict__ x, const float* __restrict__ w,
    const float* __restrict__ b, float* __restrict__ out, int D) {
  __shared__ float red[8];
  const size_t o = (size_t)blockIdx.x * D;
  ln_row(x + o, w, b, out + o, D, red);
}

// token-shift mixing: xk = xn*mk + shift(xn)*(1-mk) etc., emit bf16 operands.
__global__ __launch_bounds__(256) void mix_kernel(
    const float* __restrict__ xn,
    const float* __restrict__ tmk, const float* __restrict__ tmv,
    const float* __restrict__ tmr,
    u16* __restrict__ xk, u16* __restrict__ xv, u16* __restrict__ xr,
    int T, int D, int three) {
  const int i = blockIdx.x * blockDim.x + threadIdx.x;
  if (i >= T * D) return;
  const int t = i / D, d = i - t * D;
  const float a  = xn[i];
  const float xx = (t > 0) ? xn[i - D] : 0.f;
  const float mk = tmk[d];
  xk[i] = f2bf(a * mk + xx * (1.f - mk));
  const float mr = tmr[d];
  xr[i] = f2bf(a * mr + xx * (1.f - mr));
  if (three) {
    const float mv = tmv[d];
    xv[i] = f2bf(a * mv + xx * (1.f - mv));
  }
}

// Numerically-stabilized WKV scan: one thread per channel, serial over T.
// Latency-bound serial exp-chain -> prefetch 8 rows ahead to hide HBM latency.
__global__ __launch_bounds__(256) void wkv_kernel(
    const float* __restrict__ k, const float* __restrict__ v,
    const float* __restrict__ rpre,
    const float* __restrict__ tf_, const float* __restrict__ tdraw_,
    u16* __restrict__ out_bf16, int T, int D) {
  const int d = blockIdx.x * blockDim.x + threadIdx.x;
  if (d >= D) return;
  const float tf = tf_[d];
  const float td = -__expf(tdraw_[d]);
  float aa = 0.f, bb = 0.f, pp = -1e30f;
  for (int t = 0; t < T; ++t) {
    if (t + 8 < T) {
      __builtin_prefetch(&k[(t + 8) * D + d], 0, 1);
      __builtin_prefetch(&v[(t + 8) * D + d], 0, 1);
      __builtin_prefetch(&rpre[(t + 8) * D + d], 0, 1);
    }
    const float kt = k[t * D + d];
    const float vt = v[t * D + d];
    const float ww = tf + kt;
    const float p  = fmaxf(pp, ww);
    const float e1 = __expf(pp - p);
    const float e2 = __expf(ww - p);
    const float o  = (e1 * aa + e2 * vt) / (e1 * bb + e2);
    const float w2 = pp + td;
    const float p2 = fmaxf(w2, kt);
    const float f1 = __expf(w2 - p2);
    const float f2 = __expf(kt - p2);
    aa = f1 * aa + f2 * vt;
    bb = f1 * bb + f2;
    pp = p2;
    out_bf16[t * D + d] = f2bf(sigmoidf_(rpre[t * D + d]) * o);
  }
}

// ---------------------------------------------------------------- WMMA GEMM
// C[M,N](f32 acc) = A[M,K](bf16 row-major) x B[K,N](f32 row-major, cast->bf16)
// Double-buffered DMA pipeline:
//   A tile:  TENSOR_LOAD_TO_LDS (TDM, wave0 issues one 128x32 bf16 tile D#)
//   B tile:  raw f32 via GLOBAL_LOAD_ASYNC_TO_LDS_B128 (4 per thread)
//   convert: LDS f32 -> bf16 transposed (VALU work overlapping next tile DMA)
// epilogue modes: 0: C=acc | 1: Cb=bf16(relu(acc)^2) | 2: C+=acc
//                 3: C += sigmoid(G)*acc
#define BM 128
#define BN 128
#define BK 32
#define SA_STRIDE 32
#define SB_STRIDE 40

__global__ __launch_bounds__(256) void gemm_bf16_kernel(
    const u16* __restrict__ A, const float* __restrict__ B,
    float* __restrict__ C, u16* __restrict__ Cb, const float* __restrict__ G,
    int M, int N, int K, int mode) {
  __shared__ __align__(16) u16   sA[2][BM * SA_STRIDE];   // 2 x  8 KB (TDM dest)
  __shared__ __align__(16) float sBf[2][BK * BN];         // 2 x 16 KB (async dest)
  __shared__ __align__(16) u16   sB[2][BN * SB_STRIDE];   // 2 x 10 KB (bf16, transposed)

  const int tid  = threadIdx.x;
  const int lane = tid & 31;
  const int wave = tid >> 5;
  const int wm   = wave & 1;        // 2x4 wave grid -> each wave: 64x32 of C
  const int wn   = wave >> 1;
  const int bm   = blockIdx.y * BM;
  const int bn   = blockIdx.x * BN;

  f32x8 acc[4][2] = {};

  const int a_row = tid >> 1;           // 0..127   (fallback A staging)
  const int a_col = (tid & 1) * 16;     // 0 / 16
  const int b_row = tid >> 3;           // 0..31  (K)
  const int b_col = (tid & 7) * 16;     // 0..112 (N)
  (void)a_row; (void)a_col;

#if HAVE_TDM
  // Loop-invariant D# group1: data_size=2B, tensor 2D [M x K] elems,
  // tile [BM x BK], dim0 stride = K elements.
  i32x8 g1;
  g1[0] = 1 << 16;                                              // data_size=1 (2B)
  g1[1] = (int)(((u32)K & 0xFFFFu) << 16);                      // tensor_dim0 lo
  g1[2] = (int)((((u32)K >> 16) & 0xFFFFu) |
                (((u32)M & 0xFFFFu) << 16));                    // dim0 hi | dim1 lo
  g1[3] = (int)((((u32)M >> 16) & 0xFFFFu) | ((u32)BK << 16));  // dim1 hi | tile_dim0
  g1[4] = BM;                                                   // tile_dim1 (dim2=0)
  g1[5] = K;                                                    // dim0_stride lo
  g1[6] = 0;
  g1[7] = 0;
  i32x4 gz4 = {0, 0, 0, 0};
  i32x8 gz8 = {0, 0, 0, 0, 0, 0, 0, 0};
#endif

  // stage tile `it2` into buffer it2&1 (A via TDM, B raw f32 via async DMA)
  auto stage = [&](int it2) {
    const int kko = it2 * BK;
    const int buf = it2 & 1;
    const float* bg = B + (size_t)(kko + b_row) * N + bn + b_col;
    const u32 bd = lds_off(&sBf[buf][b_row * BN + b_col]);
    asm volatile(
        "global_load_async_to_lds_b128 %0, %4, off\n\t"
        "global_load_async_to_lds_b128 %1, %5, off\n\t"
        "global_load_async_to_lds_b128 %2, %6, off\n\t"
        "global_load_async_to_lds_b128 %3, %7, off"
        :
        : "v"(bd), "v"(bd + 16), "v"(bd + 32), "v"(bd + 48),
          "v"(bg), "v"(bg + 4), "v"(bg + 8), "v"(bg + 12)
        : "memory");
#if HAVE_TDM
    if (wave == 0) {
      const unsigned long long ga =
          (unsigned long long)(const char*)(A + (size_t)bm * K + kko);
      u32x4 g0;
      g0[0] = 1u;                                          // count=1 (valid)
      g0[1] = lds_off(&sA[buf][0]);                        // lds_addr
      g0[2] = (u32)ga;                                     // global_addr lo
      g0[3] = (u32)((ga >> 32) & 0x1FFFFFFu) | 0x80000000u; // addr hi | type=2
      __builtin_amdgcn_tensor_load_to_lds(g0, g1, gz4, gz4, gz8, 0);
    }
#else
    const u16* ag = A + (size_t)(bm + a_row) * K + kko + a_col;
    const u32 ad = lds_off(&sA[buf][a_row * SA_STRIDE + a_col]);
    asm volatile(
        "global_load_async_to_lds_b128 %0, %2, off\n\t"
        "global_load_async_to_lds_b128 %1, %3, off"
        :
        : "v"(ad), "v"(ad + 16), "v"(ag), "v"(ag + 8)
        : "memory");
#endif
  };

  // ISA 16-bit fragment geometry (wave32):
  const int ka = (lane >> 4) * 8;       // A: lanes<16 -> K 0-7/16-23 ; >=16 -> 8-15/24-31
  const int kb = (lane >> 4) * 8;       // B: lanes<16 -> K 0-7/16-23 ; >=16 -> 8-15/24-31
  const int rc = lane & 15;             // row (A) / col (B,D)

  const int steps = K / BK;
  stage(0);
  for (int it = 0; it < steps; ++it) {
    const int buf = it & 1;
    // only tile `it` is outstanding here -> wait to zero
    asm volatile("s_wait_asynccnt 0x0" ::: "memory");
#if HAVE_TDM
    if (wave == 0) __builtin_amdgcn_s_wait_tensorcnt(0);
#endif
    __syncthreads();   // every wave done reading buffers buf^1 -> safe to refill
    if (it + 1 < steps) stage(it + 1);

    // convert raw B f32 -> bf16, transposed store (overlaps next tile DMA)
    union { float4 q[4]; float f[16]; } bv;
    const float4* s4 = (const float4*)&sBf[buf][b_row * BN + b_col];
    bv.q[0] = s4[0]; bv.q[1] = s4[1]; bv.q[2] = s4[2]; bv.q[3] = s4[3];
    #pragma unroll
    for (int j = 0; j < 16; ++j)
      sB[buf][(b_col + j) * SB_STRIDE + b_row] = f2bf(bv.f[j]);
    __syncthreads();

    bf16x16 afr[4], bfr[2];
    #pragma unroll
    for (int mi = 0; mi < 4; ++mi) {
      const u16* p = &sA[buf][(wm * 64 + mi * 16 + rc) * SA_STRIDE + ka];
      union { uint4 u[2]; bf16x16 v; } f;
      f.u[0] = *(const uint4*)p;
      f.u[1] = *(const uint4*)(p + 16);
      afr[mi] = f.v;
    }
    #pragma unroll
    for (int ni = 0; ni < 2; ++ni) {
      const u16* p = &sB[buf][(wn * 32 + ni * 16 + rc) * SB_STRIDE + kb];
      union { uint4 u[2]; bf16x16 v; } f;
      f.u[0] = *(const uint4*)p;
      f.u[1] = *(const uint4*)(p + 16);
      bfr[ni] = f.v;
    }
    #pragma unroll
    for (int mi = 0; mi < 4; ++mi)
      #pragma unroll
      for (int ni = 0; ni < 2; ++ni)
        acc[mi][ni] = __builtin_amdgcn_wmma_f32_16x16x32_bf16(
            false, afr[mi], false, bfr[ni], (short)0, acc[mi][ni], false, false);
  }

  // epilogue: D layout = VGPR j -> M = (lane>=16)*8 + j, N = lane&15
  #pragma unroll
  for (int mi = 0; mi < 4; ++mi) {
    #pragma unroll
    for (int ni = 0; ni < 2; ++ni) {
      #pragma unroll
      for (int j = 0; j < 8; ++j) {
        const int row = bm + wm * 64 + mi * 16 + (lane >> 4) * 8 + j;
        const int col = bn + wn * 32 + ni * 16 + rc;
        const size_t idx = (size_t)row * N + col;
        const float v = acc[mi][ni][j];
        if (mode == 0) {
          C[idx] = v;
        } else if (mode == 1) {
          const float r = fmaxf(v, 0.f);
          Cb[idx] = f2bf(r * r);
        } else if (mode == 2) {
          C[idx] += v;
        } else {
          C[idx] += sigmoidf_(G[idx]) * v;
        }
      }
    }
  }
}

// logits = head @ y : pure bandwidth (131 MB), one wave per row, float4 loads.
__global__ __launch_bounds__(256) void head_kernel(
    const float* __restrict__ headw, const float* __restrict__ y,
    float* __restrict__ out, int V, int D) {
  const int wid  = (int)((blockIdx.x * blockDim.x + threadIdx.x) >> 5);
  const int lane = threadIdx.x & 31;
  if (wid >= V) return;
  const float4* row = (const float4*)(headw + (size_t)wid * D);
  const float4* yv  = (const float4*)y;
  float s = 0.f;
  for (int j = lane; j < D / 4; j += 32) {
    const float4 a = row[j];
    const float4 b = yv[j];
    s += a.x * b.x + a.y * b.y + a.z * b.z + a.w * b.w;
  }
  #pragma unroll
  for (int o = 16; o > 0; o >>= 1) s += __shfl_down(s, o, 32);
  if (lane == 0) out[wid] = s;
}

// ---------------------------------------------------------------- launch
extern "C" void kernel_launch(void* const* d_in, const int* in_sizes, int n_in,
                              void* d_out, int out_size, void* d_ws, size_t ws_size,
                              hipStream_t stream) {
  const int T = 1024, D = 1024, F = 4096, V = 32000, L = 12;
  (void)in_sizes; (void)n_in; (void)out_size; (void)ws_size;

  const int*   tokens  = (const int*)  d_in[0];
  const float* emb     = (const float*)d_in[1];
  const float* ln0_w   = (const float*)d_in[2];
  const float* ln0_b   = (const float*)d_in[3];
  const float* ln1_w   = (const float*)d_in[4];
  const float* ln1_b   = (const float*)d_in[5];
  const float* ln2_w   = (const float*)d_in[6];
  const float* ln2_b   = (const float*)d_in[7];
  const float* att_tmk = (const float*)d_in[8];
  const float* att_tmv = (const float*)d_in[9];
  const float* att_tmr = (const float*)d_in[10];
  const float* att_tf  = (const float*)d_in[11];
  const float* att_td  = (const float*)d_in[12];
  const float* att_kw  = (const float*)d_in[13];
  const float* att_vw  = (const float*)d_in[14];
  const float* att_rw  = (const float*)d_in[15];
  const float* att_ow  = (const float*)d_in[16];
  const float* ffn_tmk = (const float*)d_in[17];
  const float* ffn_tmr = (const float*)d_in[18];
  const float* ffn_kw  = (const float*)d_in[19];
  const float* ffn_vw  = (const float*)d_in[20];
  const float* ffn_rw  = (const float*)d_in[21];
  const float* lnout_w = (const float*)d_in[22];
  const float* lnout_b = (const float*)d_in[23];
  const float* headw   = (const float*)d_in[24];
  float* logits = (float*)d_out;

  char* ws = (char*)d_ws;
  size_t off = 0;
  auto carve = [&](size_t bytes) -> void* {
    void* p = ws + off;
    off += (bytes + 255) & ~(size_t)255;
    return p;
  };
  float* x     = (float*)carve((size_t)T * D * 4);
  float* xn    = (float*)carve((size_t)T * D * 4);
  u16*   xk    = (u16*)  carve((size_t)T * D * 2);
  u16*   xv    = (u16*)  carve((size_t)T * D * 2);
  u16*   xr    = (u16*)  carve((size_t)T * D * 2);
  float* kbuf  = (float*)carve((size_t)T * D * 4);
  float* vbuf  = (float*)carve((size_t)T * D * 4);
  float* rpre  = (float*)carve((size_t)T * D * 4);
  u16*   ga    = (u16*)  carve((size_t)T * D * 2);
  float* r2pre = (float*)carve((size_t)T * D * 4);
  u16*   kkb   = (u16*)  carve((size_t)T * F * 2);
  float* yvec  = (float*)carve((size_t)D * 4);

  embed_ln0_kernel<<<T, 256, 0, stream>>>(tokens, emb, ln0_w, ln0_b, x, D);

  const dim3 blk(256);
  const dim3 gDD(D / BN, T / BM);   // N=1024 tiles x M=1024 tiles
  const dim3 gDF(F / BN, T / BM);   // N=4096
  const int mixBlocks = (T * D + 255) / 256;

  for (int l = 0; l < L; ++l) {
    const size_t od  = (size_t)l * D;
    const size_t odd = (size_t)l * D * D;
    const size_t odf = (size_t)l * D * F;
    // ---- time-mixing
    ln_rows_kernel<<<T, blk, 0, stream>>>(x, ln1_w + od, ln1_b + od, xn, D);
    mix_kernel<<<mixBlocks, blk, 0, stream>>>(xn, att_tmk + od, att_tmv + od,
                                              att_tmr + od, xk, xv, xr, T, D, 1);
    gemm_bf16_kernel<<<gDD, blk, 0, stream>>>(xk, att_kw + odd, kbuf, nullptr, nullptr, T, D, D, 0);
    gemm_bf16_kernel<<<gDD, blk, 0, stream>>>(xv, att_vw + odd, vbuf, nullptr, nullptr, T, D, D, 0);
    gemm_bf16_kernel<<<gDD, blk, 0, stream>>>(xr, att_rw + odd, rpre, nullptr, nullptr, T, D, D, 0);
    wkv_kernel<<<D / 256, blk, 0, stream>>>(kbuf, vbuf, rpre, att_tf + od, att_td + od, ga, T, D);
    gemm_bf16_kernel<<<gDD, blk, 0, stream>>>(ga, att_ow + odd, x, nullptr, nullptr, T, D, D, 2);
    // ---- channel-mixing
    ln_rows_kernel<<<T, blk, 0, stream>>>(x, ln2_w + od, ln2_b + od, xn, D);
    mix_kernel<<<mixBlocks, blk, 0, stream>>>(xn, ffn_tmk + od, ffn_tmk + od,
                                              ffn_tmr + od, xk, xv, xr, T, D, 0);
    gemm_bf16_kernel<<<gDD, blk, 0, stream>>>(xr, ffn_rw + odd, r2pre, nullptr, nullptr, T, D, D, 0);
    gemm_bf16_kernel<<<gDF, blk, 0, stream>>>(xk, ffn_kw + odf, nullptr, kkb, nullptr, T, F, D, 1);
    gemm_bf16_kernel<<<gDD, blk, 0, stream>>>(kkb, ffn_vw + odf, x, nullptr, r2pre, T, D, F, 3);
  }

  ln_rows_kernel<<<1, blk, 0, stream>>>(x + (size_t)(T - 1) * D, lnout_w, lnout_b, yvec, D);
  head_kernel<<<(V * 32 + 255) / 256, blk, 0, stream>>>(headw, yvec, logits, V, D);
}
